// RelModel_721554506169
// MI455X (gfx1250) — matrix operand order repaired and checked by reading, not compiled, not run on hardware
//
#include <hip/hip_runtime.h>
#include <hip/hip_bf16.h>

typedef __attribute__((ext_vector_type(16))) _Float16 v16h;
typedef __attribute__((ext_vector_type(8)))  float    v8f;

// ---------------------------------------------------------------------------
// Problem constants
// ---------------------------------------------------------------------------
constexpr int NROW  = 4096;   // objects
constexpr int HID   = 256;
constexpr int EMB   = 200;
constexpr int IN1   = 456;    // EMB + HID
constexpr int INT1  = 228;
constexpr int IN2   = 256;
constexpr int INT2  = 128;
constexpr int OUT2  = 8192;

constexpr int IN1P  = 512;    // IN1 padded (lda for inp/o1, N-pad for W0)
constexpr int K1P   = 480;    // IN1 padded to mult of 32 (K for RE1 qkv / W1)
constexpr int INT1P = 256;    // INT1 padded (N for qkv, K for W0/attn)

// ---------------------------------------------------------------------------
// Fragment loaders (CDNA5 wave32 WMMA VGPR layouts, 16-bit operands)
//   A 16x32 (MxK): lane<16 holds K {k..k+7, k+16..k+23}; lane>=16 holds +8
//   B 32x16 (KxN): lane = column; lane<16 holds K {k..k+7, k+16..k+23}; lane>=16 holds +8
// ---------------------------------------------------------------------------
__device__ __forceinline__ v16h ld_frag2(const _Float16* __restrict__ p, int second_off) {
    union { uint4 u[2]; v16h h; } x;
    x.u[0] = *(const uint4*)(p);
    x.u[1] = *(const uint4*)(p + second_off);
    return x.h;
}

#define WMMA_F16(A, B, C) \
    __builtin_amdgcn_wmma_f32_16x16x32_f16(false, (A), false, (B), (short)0, (C), false, false)

// ---------------------------------------------------------------------------
// NT GEMM:  C[M,Ncols] = A[M,K](f16,row-major) * B[Ncols,K](f16,row-major)^T
//   + optional bias (f32, valid for n < Nact), + optional f16 residual,
//   writing f16 (optionally transposed) and/or f32 outputs.
// Block = 128 threads (4 waves). Block tile 64x64; wave tile 32x32 (2x2 frags).
// M, Ncols multiples of 64; K multiple of 32; all lds multiples of 8.
// ---------------------------------------------------------------------------
template<bool F16OUT, bool F32OUT, bool TRANS, bool BIAS, bool RESID>
__global__ __launch_bounds__(128) void wmma_gemm_nt(
    const _Float16* __restrict__ A, int lda,
    const _Float16* __restrict__ B, int ldb,
    const float*    __restrict__ bias, int Nact,
    const _Float16* __restrict__ resid, int ldr,
    _Float16*       __restrict__ Ch, int ldch,
    float*          __restrict__ Cf, int ldcf,
    int K)
{
    const int lane = threadIdx.x & 31;
    const int wave = threadIdx.x >> 5;
    const int mBase = blockIdx.y * 64 + (wave & 1) * 32;
    const int nBase = blockIdx.x * 64 + (wave >> 1) * 32;

    v8f acc[2][2] = {};

    const int aRow0 = mBase + (lane & 15);
    const int bCol0 = nBase + (lane & 15);
    const int aKoff = (lane >> 4) << 3;    // 0 or 8
    const int bKoff = (lane >> 4) << 3;    // 0 or 8

    const _Float16* aP0 = A + (size_t)aRow0 * lda + aKoff;
    const _Float16* aP1 = aP0 + (size_t)16 * lda;
    const _Float16* bP0 = B + (size_t)bCol0 * ldb + bKoff;
    const _Float16* bP1 = bP0 + (size_t)16 * ldb;

    for (int k = 0; k < K; k += 32) {
        v16h a0 = ld_frag2(aP0 + k, 16);
        v16h a1 = ld_frag2(aP1 + k, 16);
        v16h b0 = ld_frag2(bP0 + k, 16);
        v16h b1 = ld_frag2(bP1 + k, 16);
        acc[0][0] = WMMA_F16(a0, b0, acc[0][0]);
        acc[0][1] = WMMA_F16(a0, b1, acc[0][1]);
        acc[1][0] = WMMA_F16(a1, b0, acc[1][0]);
        acc[1][1] = WMMA_F16(a1, b1, acc[1][1]);
    }

    // D layout: VGPR r, lanes 0-15 -> M = r, lanes 16-31 -> M = 8 + r; N = lane&15
    const int mOut0 = mBase + ((lane >> 4) << 3);
    const int nOut0 = nBase + (lane & 15);
    #pragma unroll
    for (int mi = 0; mi < 2; ++mi) {
        #pragma unroll
        for (int ni = 0; ni < 2; ++ni) {
            const int n = nOut0 + ni * 16;
            float bv = 0.f;
            if (BIAS) bv = (n < Nact) ? bias[n] : 0.f;
            #pragma unroll
            for (int r = 0; r < 8; ++r) {
                const int m = mOut0 + mi * 16 + r;
                float v = acc[mi][ni][r] + bv;
                if (RESID) v += (float)resid[(size_t)m * ldr + n];
                if (F32OUT) Cf[(size_t)m * ldcf + n] = v;
                if (F16OUT) {
                    if (TRANS) Ch[(size_t)n * ldch + m] = (_Float16)v;
                    else       Ch[(size_t)m * ldch + n] = (_Float16)v;
                }
            }
        }
    }
}

template<bool F16OUT, bool F32OUT, bool TRANS, bool BIAS, bool RESID>
static void gemm(hipStream_t s,
                 const _Float16* A, int lda, const _Float16* B, int ldb,
                 const float* bias, int Nact, const _Float16* resid, int ldr,
                 _Float16* Ch, int ldch, float* Cf, int ldcf,
                 int M, int Ncols, int K)
{
    dim3 g(Ncols / 64, M / 64), b(128);
    wmma_gemm_nt<F16OUT, F32OUT, TRANS, BIAS, RESID>
        <<<g, b, 0, s>>>(A, lda, B, ldb, bias, Nact, resid, ldr, Ch, ldch, Cf, ldcf, K);
}

// ---------------------------------------------------------------------------
// Prep: inpH[m, 0:200] = embed2[preds[m]]; [200:456] = obj_ctx[m]; rest 0. f16.
// ---------------------------------------------------------------------------
__global__ void prep_input(const float* __restrict__ ctx, const int* __restrict__ preds,
                           const float* __restrict__ emb, _Float16* __restrict__ inpH)
{
    int idx = blockIdx.x * 256 + threadIdx.x;      // NROW * IN1P total
    int m = idx >> 9;                              // / 512
    int c = idx & 511;
    float v = 0.f;
    if (c < EMB)      v = emb[preds[m] * EMB + c];
    else if (c < IN1) v = ctx[m * HID + (c - EMB)];
    inpH[idx] = (_Float16)v;
}

// ---------------------------------------------------------------------------
// Weight pad + f32->f16: dst[NP,KP] = src[N,K] zero-padded
// ---------------------------------------------------------------------------
__global__ void pad_w(const float* __restrict__ src, _Float16* __restrict__ dst,
                      int N, int K, int NP, int KP)
{
    int idx = blockIdx.x * 256 + threadIdx.x;
    if (idx >= NP * KP) return;
    int n = idx / KP, k = idx - n * KP;
    dst[idx] = (_Float16)((n < N && k < K) ? src[n * K + k] : 0.f);
}

// ---------------------------------------------------------------------------
// Row softmax: S (f32 [rows, n]) -> P (f16 [rows, n]); one 256-thread block/row
// ---------------------------------------------------------------------------
__global__ void softmax_rows(const float* __restrict__ S, _Float16* __restrict__ P, int n)
{
    const int row = blockIdx.x, tid = threadIdx.x;
    const float* s = S + (size_t)row * n;
    _Float16*    p = P + (size_t)row * n;
    __shared__ float red[256];

    float mx = -3.4e38f;
    for (int i = tid; i < n; i += 256) mx = fmaxf(mx, s[i]);
    red[tid] = mx; __syncthreads();
    for (int st = 128; st > 0; st >>= 1) { if (tid < st) red[tid] = fmaxf(red[tid], red[tid + st]); __syncthreads(); }
    mx = red[0]; __syncthreads();

    float sum = 0.f;
    for (int i = tid; i < n; i += 256) sum += __expf(s[i] - mx);
    red[tid] = sum; __syncthreads();
    for (int st = 128; st > 0; st >>= 1) { if (tid < st) red[tid] += red[tid + st]; __syncthreads(); }
    const float inv = 1.f / red[0];

    for (int i = tid; i < n; i += 256) p[i] = (_Float16)(__expf(s[i] - mx) * inv);
}

// ---------------------------------------------------------------------------
// Launch
// ---------------------------------------------------------------------------
extern "C" void kernel_launch(void* const* d_in, const int* in_sizes, int n_in,
                              void* d_out, int out_size, void* d_ws, size_t ws_size,
                              hipStream_t stream)
{
    (void)in_sizes; (void)n_in; (void)out_size; (void)ws_size;

    const float* obj_ctx = (const float*)d_in[0];
    const int*   preds   = (const int*)  d_in[1];
    const float* embed2  = (const float*)d_in[2];
    const float* r1_Wq = (const float*)d_in[3];  const float* r1_bq = (const float*)d_in[4];
    const float* r1_Wk = (const float*)d_in[5];  const float* r1_bk = (const float*)d_in[6];
    const float* r1_Wv = (const float*)d_in[7];  const float* r1_bv = (const float*)d_in[8];
    const float* r1_W0 = (const float*)d_in[9];  const float* r1_b0 = (const float*)d_in[10];
    const float* r1_W1 = (const float*)d_in[11]; const float* r1_b1 = (const float*)d_in[12];
    const float* r2_Wq = (const float*)d_in[13]; const float* r2_bq = (const float*)d_in[14];
    const float* r2_Wk = (const float*)d_in[15]; const float* r2_bk = (const float*)d_in[16];
    const float* r2_Wv = (const float*)d_in[17]; const float* r2_bv = (const float*)d_in[18];
    const float* r2_W0 = (const float*)d_in[19]; const float* r2_b0 = (const float*)d_in[20];
    const float* r2_W1 = (const float*)d_in[21]; const float* r2_b1 = (const float*)d_in[22];

    float* outCtx = (float*)d_out;                       // [4096, 256]
    float* outRep = outCtx + (size_t)NROW * HID;         // [4096, 8192]

    // ---- workspace layout (256B aligned) ----
    char* ws = (char*)d_ws;
    size_t off = 0;
    auto alloc = [&](size_t bytes) { void* p = ws + off; off = (off + bytes + 255) & ~(size_t)255; return p; };

    _Float16* inpH  = (_Float16*)alloc((size_t)NROW * IN1P * 2);
    _Float16* Wq1H  = (_Float16*)alloc((size_t)INT1P * K1P * 2);
    _Float16* Wk1H  = (_Float16*)alloc((size_t)INT1P * K1P * 2);
    _Float16* Wv1H  = (_Float16*)alloc((size_t)INT1P * K1P * 2);
    _Float16* W01H  = (_Float16*)alloc((size_t)IN1P * INT1P * 2);
    _Float16* W11H  = (_Float16*)alloc((size_t)HID * K1P * 2);
    _Float16* Wq2H  = (_Float16*)alloc((size_t)INT2 * IN2 * 2);
    _Float16* Wk2H  = (_Float16*)alloc((size_t)INT2 * IN2 * 2);
    _Float16* Wv2H  = (_Float16*)alloc((size_t)INT2 * IN2 * 2);
    _Float16* W02H  = (_Float16*)alloc((size_t)IN2 * INT2 * 2);
    _Float16* W12H  = (_Float16*)alloc((size_t)OUT2 * IN2 * 2);
    _Float16* qH    = (_Float16*)alloc((size_t)NROW * INT1P * 2);
    _Float16* kH    = (_Float16*)alloc((size_t)NROW * INT1P * 2);
    _Float16* vT    = (_Float16*)alloc((size_t)INT1P * NROW * 2);
    float*    S     = (float*)   alloc((size_t)NROW * NROW * 4);
    _Float16* P     = (_Float16*)alloc((size_t)NROW * NROW * 2);
    _Float16* AVh   = (_Float16*)alloc((size_t)NROW * INT1P * 2);
    _Float16* o1H   = (_Float16*)alloc((size_t)NROW * IN1P * 2);
    _Float16* t2H   = (_Float16*)alloc((size_t)NROW * HID * 2);
    _Float16* q2H   = (_Float16*)alloc((size_t)NROW * INT2 * 2);
    _Float16* k2H   = (_Float16*)alloc((size_t)NROW * INT2 * 2);
    _Float16* v2T   = (_Float16*)alloc((size_t)INT2 * NROW * 2);
    _Float16* AV2h  = (_Float16*)alloc((size_t)NROW * INT2 * 2);
    _Float16* ecH   = (_Float16*)alloc((size_t)NROW * HID * 2);

    // ---- stage 0: input assembly + weight conversion ----
    prep_input<<<(NROW * IN1P) / 256, 256, 0, stream>>>(obj_ctx, preds, embed2, inpH);

    auto padw = [&](const float* s, _Float16* d, int N, int K, int NP, int KP) {
        int tot = NP * KP;
        pad_w<<<(tot + 255) / 256, 256, 0, stream>>>(s, d, N, K, NP, KP);
    };
    padw(r1_Wq, Wq1H, INT1, IN1, INT1P, K1P);
    padw(r1_Wk, Wk1H, INT1, IN1, INT1P, K1P);
    padw(r1_Wv, Wv1H, INT1, IN1, INT1P, K1P);
    padw(r1_W0, W01H, IN1, INT1, IN1P, INT1P);
    padw(r1_W1, W11H, HID, IN1, HID, K1P);
    padw(r2_Wq, Wq2H, INT2, IN2, INT2, IN2);
    padw(r2_Wk, Wk2H, INT2, IN2, INT2, IN2);
    padw(r2_Wv, Wv2H, INT2, IN2, INT2, IN2);
    padw(r2_W0, W02H, IN2, INT2, IN2, INT2);
    padw(r2_W1, W12H, OUT2, IN2, OUT2, IN2);

    // ---- RE1 ----
    gemm<1,0,0,1,0>(stream, inpH, IN1P, Wq1H, K1P, r1_bq, INT1, nullptr, 0,
                    qH, INT1P, nullptr, 0, NROW, INT1P, K1P);
    gemm<1,0,0,1,0>(stream, inpH, IN1P, Wk1H, K1P, r1_bk, INT1, nullptr, 0,
                    kH, INT1P, nullptr, 0, NROW, INT1P, K1P);
    gemm<1,0,1,1,0>(stream, inpH, IN1P, Wv1H, K1P, r1_bv, INT1, nullptr, 0,
                    vT, NROW, nullptr, 0, NROW, INT1P, K1P);           // v stored transposed
    gemm<0,1,0,0,0>(stream, qH, INT1P, kH, INT1P, nullptr, 0, nullptr, 0,
                    nullptr, 0, S, NROW, NROW, NROW, INT1P);           // scores
    softmax_rows<<<NROW, 256, 0, stream>>>(S, P, NROW);
    gemm<1,0,0,0,0>(stream, P, NROW, vT, NROW, nullptr, 0, nullptr, 0,
                    AVh, INT1P, nullptr, 0, NROW, INT1P, NROW);        // attn @ v
    gemm<1,0,0,1,1>(stream, AVh, INT1P, W01H, INT1P, r1_b0, IN1, inpH, IN1P,
                    o1H, IN1P, nullptr, 0, NROW, IN1P, INT1P);         // o1 = inp + AV@W0.T + b0
    gemm<1,0,0,1,0>(stream, o1H, IN1P, W11H, K1P, r1_b1, HID, nullptr, 0,
                    t2H, HID, nullptr, 0, NROW, HID, K1P);             // tmp2

    // ---- RE2 ----
    gemm<1,0,0,1,0>(stream, t2H, HID, Wq2H, IN2, r2_bq, INT2, nullptr, 0,
                    q2H, INT2, nullptr, 0, NROW, INT2, IN2);
    gemm<1,0,0,1,0>(stream, t2H, HID, Wk2H, IN2, r2_bk, INT2, nullptr, 0,
                    k2H, INT2, nullptr, 0, NROW, INT2, IN2);
    gemm<1,0,1,1,0>(stream, t2H, HID, Wv2H, IN2, r2_bv, INT2, nullptr, 0,
                    v2T, NROW, nullptr, 0, NROW, INT2, IN2);
    gemm<0,1,0,0,0>(stream, q2H, INT2, k2H, INT2, nullptr, 0, nullptr, 0,
                    nullptr, 0, S, NROW, NROW, NROW, INT2);
    softmax_rows<<<NROW, 256, 0, stream>>>(S, P, NROW);
    gemm<1,0,0,0,0>(stream, P, NROW, v2T, NROW, nullptr, 0, nullptr, 0,
                    AV2h, INT2, nullptr, 0, NROW, INT2, NROW);
    gemm<1,1,0,1,1>(stream, AV2h, INT2, W02H, INT2, r2_b0, HID, t2H, HID,
                    ecH, HID, outCtx, HID, NROW, HID, INT2);           // edge_ctx (f32 out + f16 feed)
    gemm<0,1,0,1,0>(stream, ecH, HID, W12H, IN2, r2_b1, OUT2, nullptr, 0,
                    nullptr, 0, outRep, OUT2, NROW, OUT2, IN2);        // edge_rep
}
